// BaseBlock_50319836840391
// MI455X (gfx1250) — matrix operand order repaired and checked by reading, not compiled, not run on hardware
//
#include <hip/hip_runtime.h>

// ---------------------------------------------------------------------------
// Types for CDNA5 WMMA
// ---------------------------------------------------------------------------
typedef __attribute__((ext_vector_type(16))) _Float16 v16h;
typedef __attribute__((ext_vector_type(8)))  float    v8f;
typedef __attribute__((ext_vector_type(2)))  _Float16 h2;
typedef __attribute__((ext_vector_type(4)))  _Float16 h4;

#define HW_   65536
#define CHW_  12582912   // 192*65536
#define CD_   192
#define XSTR  48         // LDS row stride in halfs (96 B: 16B-aligned rows)

// ---------------------------------------------------------------------------
// Scratch layout (bytes)
// ---------------------------------------------------------------------------
constexpr size_t OFF_QKV = 0;                          // f16 [131072][576]
constexpr size_t OFF_B   = 150994944;                  // f16 [131072][192] (xh / attn_out / ln2h)
constexpr size_t OFF_C   = OFF_B   + 50331648;         // f32 y_attn BCHW  OR f16 [131072][384]
constexpr size_t OFF_MU  = OFF_C   + 100663296;        // f32 [131072]
constexpr size_t OFF_RS  = OFF_MU  + 524288;           // f32 [131072]
constexpr size_t OFF_GAP = OFF_RS  + 524288;           // f32 [384]
constexpr size_t OFF_SCL = OFF_GAP + 2048;             // f32 [384]
constexpr size_t OFF_WQ  = OFF_SCL + 2048;             // f16 qkv_w  192*576
constexpr size_t OFF_WP  = OFF_WQ  + 221184;           // f16 proj_w 192*192
constexpr size_t OFF_W1  = OFF_WP  + 73728;            // f16 fc1_w  192*384
constexpr size_t OFF_W2  = OFF_W1  + 147456;           // f16 fc2_w  384*192

// ---------------------------------------------------------------------------
// Fragment helpers (A/B: this code's contiguous k order; C per CDNA5 ISA §7.12.2)
// A 16x32 f16: lane m=lane&15 holds row m; laneHalf selects K offset +16;
//   element i -> K = i + 16*laneHalf
// B 32x16 f16: lane n=lane&15 holds col n; laneHalf selects K offset +16;
//   element i -> K = i + 16*laneHalf
// C 16x16 f32: element r -> row r + 8*laneHalf, col lane&15
// ---------------------------------------------------------------------------
__device__ __forceinline__ v8f wmma_f16(v16h a, v16h b, v8f c) {
    return __builtin_amdgcn_wmma_f32_16x16x32_f16(false, a, false, b, (short)0, c,
                                                  false, false);
}

__device__ __forceinline__ v16h frag_rowA(const _Float16* rowp, int lh) {
    v16h f;
#pragma unroll
    for (int j = 0; j < 8; ++j) {
        int i = 2 * j;
        int k = i + (lh << 4);
        h2 t = *(const h2*)(rowp + k);
        f[i] = t.x; f[i + 1] = t.y;
    }
    return f;
}

__device__ __forceinline__ v16h frag_rowB(const _Float16* colp, int lh) {
    v16h f;
#pragma unroll
    for (int j = 0; j < 8; ++j) {
        int i = 2 * j;
        int k = i + (lh << 4);
        h2 t = *(const h2*)(colp + k);
        f[i] = t.x; f[i + 1] = t.y;
    }
    return f;
}

__device__ __forceinline__ v16h frag_colB(const _Float16* base, int ld, int lh) {
    v16h f;
#pragma unroll
    for (int i = 0; i < 16; ++i) f[i] = base[(long)(i + (lh << 4)) * ld];
    return f;
}

__device__ __forceinline__ int reg8(int g) { return g < 248 ? 0 : (g < 252 ? 1 : 2); }

// ---------------------------------------------------------------------------
// f32 -> f16 weight conversion
// ---------------------------------------------------------------------------
__global__ void __launch_bounds__(256)
cvt_f16_kernel(const float* __restrict__ src, _Float16* __restrict__ dst, int n) {
    int i = blockIdx.x * 256 + threadIdx.x;
    if (i < n) dst[i] = (_Float16)src[i];
}

// ---------------------------------------------------------------------------
// LayerNorm stats over channels per pixel (BCHW)
// ---------------------------------------------------------------------------
__global__ void __launch_bounds__(256)
ln_stats_kernel(const float* __restrict__ x, float* __restrict__ mu,
                float* __restrict__ rs) {
    int p = blockIdx.x * 256 + threadIdx.x;          // 0..131071
    int b = p >> 16, s = p & (HW_ - 1);
    const float* px = x + (long)b * CHW_ + s;
    float sum = 0.f, sq = 0.f;
    for (int c = 0; c < CD_; ++c) {
        float v = px[(long)c << 16];
        sum += v; sq += v * v;
    }
    float m = sum * (1.f / CD_);
    float var = sq * (1.f / CD_) - m * m;
    mu[p] = m;
    rs[p] = rsqrtf(var + 1e-6f);
}

// ---------------------------------------------------------------------------
// LN1 + raw-reshape roll + window partition -> f16 [B_][64][192]
// ---------------------------------------------------------------------------
__global__ void __launch_bounds__(256)
ln1_window_kernel(const float* __restrict__ x, const float* __restrict__ mu,
                  const float* __restrict__ rs, const float* __restrict__ w,
                  const float* __restrict__ bb, _Float16* __restrict__ xh) {
    int e = blockIdx.x * 256 + threadIdx.x;          // < 25,165,824
    int bw = e / 12288;
    int rem = e - bw * 12288;
    int n = rem / CD_;
    int ch = rem - n * CD_;
    int b = bw >> 10, widx = bw & 1023;
    int gh = (widx >> 5) * 8 + (n >> 3);
    int gw = (widx & 31) * 8 + (n & 7);
    int p = ch * HW_ + gh * 256 + gw;                // BCHW view of rolled tensor
    int h1 = p / 49152;                              // raw (H,W,C) reinterpretation
    int r2 = p - h1 * 49152;
    int w1 = r2 / CD_;
    int c1 = r2 - w1 * CD_;
    int q = ((h1 + 4) & 255) * 49152 + ((w1 + 4) & 255) * CD_ + c1;
    int co = q >> 16, s = q & (HW_ - 1);
    int pix = b * HW_ + s;
    float v = (x[(long)b * CHW_ + q] - mu[pix]) * rs[pix] * w[co] + bb[co];
    xh[e] = (_Float16)v;
}

// ---------------------------------------------------------------------------
// LN2 -> f16 [pixel][192] (BHWC token layout for MLP)
// ---------------------------------------------------------------------------
__global__ void __launch_bounds__(256)
ln2_apply_kernel(const float* __restrict__ xo, const float* __restrict__ mu,
                 const float* __restrict__ rs, const float* __restrict__ w,
                 const float* __restrict__ bb, _Float16* __restrict__ outh) {
    int e = blockIdx.x * 256 + threadIdx.x;
    int pix = e / CD_;
    int c = e - pix * CD_;
    int b = pix >> 16, s = pix & (HW_ - 1);
    float v = (xo[(long)b * CHW_ + (long)c * HW_ + s] - mu[pix]) * rs[pix] * w[c] + bb[c];
    outh[e] = (_Float16)v;
}

// ---------------------------------------------------------------------------
// Tiled WMMA GEMM: X[f16, M x KDIM] @ W[f16, KDIM x NDIM] + bias
// 128x64 block tile, 8 waves (4 M-strips of 32 x 2 N-halves of 32),
// each wave: 2 A-frags x 2 B-frags -> 4 v_wmma per K-step.
// X tile staged with GLOBAL_LOAD_ASYNC_TO_LDS_B128 (ASYNCcnt-tracked).
// EPI: 0 = store f16 (qkv); 1 = exact GELU + f16 (fc1);
//      2 = window_reverse scatter f32 (proj), LDS-transposed coalesced;
//      3 = scatter-add BCHW f32 (fc2), LDS-transposed coalesced.
// ---------------------------------------------------------------------------
template <int KDIM, int NDIM, int EPI>
__global__ void __launch_bounds__(256)
gemm_kernel(const _Float16* __restrict__ X, const _Float16* __restrict__ W,
            const float* __restrict__ bias, float* __restrict__ outF,
            _Float16* __restrict__ outH) {
    __shared__ __align__(16) _Float16 Xs[128 * XSTR];
    __shared__ __align__(16) _Float16 Wst[64 * XSTR];
    __shared__ float tbuf[(EPI >= 2) ? 8 * 32 * 33 : 1];

    const int tid = threadIdx.x;
    const int wave = tid >> 5, lane = tid & 31;
    const int lh = (lane >> 4) & 1, lm = lane & 15;
    const int wm = wave & 3, wn = wave >> 2;
    const long row0 = (long)blockIdx.x * 128;
    const int col0 = blockIdx.y * 64;

    const v8f vz = {0.f, 0.f, 0.f, 0.f, 0.f, 0.f, 0.f, 0.f};
    v8f acc[2][2] = {{vz, vz}, {vz, vz}};

    for (int k0 = 0; k0 < KDIM; k0 += 32) {
        // --- async stage X tile 128x32 halfs (16 B per lane, 2 issues) ---
        {
            int idx = tid;
#pragma unroll
            for (int ii = 0; ii < 2; ++ii, idx += 256) {
                int r = idx >> 2, ck = (idx & 3) << 3;
                unsigned lds = (unsigned)(size_t)&Xs[r * XSTR + ck];
                unsigned long long ga =
                    (unsigned long long)(size_t)&X[(row0 + r) * KDIM + k0 + ck];
                asm volatile("global_load_async_to_lds_b128 %0, %1, off"
                             :: "v"(lds), "v"(ga) : "memory");
            }
        }
        // --- stage W tile 32x64 transposed -> Wst[col][k] ---
        for (int idx = tid; idx < 512; idx += 256) {
            int r = idx >> 4, c4 = (idx & 15) << 2;
            h4 w4 = *(const h4*)&W[(long)(k0 + r) * NDIM + col0 + c4];
            Wst[(c4 + 0) * XSTR + r] = w4.x;
            Wst[(c4 + 1) * XSTR + r] = w4.y;
            Wst[(c4 + 2) * XSTR + r] = w4.z;
            Wst[(c4 + 3) * XSTR + r] = w4.w;
        }
        asm volatile("s_wait_asynccnt 0x0" ::: "memory");
        __syncthreads();

        v16h a0 = frag_rowA(&Xs[(wm * 32 + lm) * XSTR], lh);
        v16h a1 = frag_rowA(&Xs[(wm * 32 + 16 + lm) * XSTR], lh);
        v16h b0 = frag_rowB(&Wst[(wn * 32 + lm) * XSTR], lh);
        v16h b1 = frag_rowB(&Wst[(wn * 32 + 16 + lm) * XSTR], lh);
        acc[0][0] = wmma_f16(a0, b0, acc[0][0]);
        acc[0][1] = wmma_f16(a0, b1, acc[0][1]);
        acc[1][0] = wmma_f16(a1, b0, acc[1][0]);
        acc[1][1] = wmma_f16(a1, b1, acc[1][1]);
        __syncthreads();
    }

    // ----------------------------- epilogue --------------------------------
    float* wbuf = &tbuf[(EPI >= 2) ? wave * (32 * 33) : 0];
#pragma unroll
    for (int tm = 0; tm < 2; ++tm) {
#pragma unroll
        for (int tn = 0; tn < 2; ++tn) {
#pragma unroll
            for (int rr = 0; rr < 8; ++rr) {
                int lrow = tm * 16 + rr + lh * 8;
                int lcol = tn * 16 + lm;
                int col = col0 + wn * 32 + lcol;
                float v = acc[tm][tn][rr] + bias[col];
                if constexpr (EPI == 0) {
                    int row = (int)row0 + wm * 32 + lrow;
                    outH[(long)row * NDIM + col] = (_Float16)v;
                } else if constexpr (EPI == 1) {
                    int row = (int)row0 + wm * 32 + lrow;
                    float g = 0.5f * v * (1.0f + erff(v * 0.7071067811865475f));
                    outH[(long)row * NDIM + col] = (_Float16)g;
                } else {
                    wbuf[lcol * 33 + lrow] = v;   // transpose via LDS
                }
            }
        }
    }
    if constexpr (EPI == 2) {
        // lane owns one window token; 4 runs of 8 contiguous pixels per store
        int row = (int)row0 + wm * 32 + lane;
        int bw = row >> 6, n = row & 63;
        int b = bw >> 10, widx = bw & 1023;
        int gh = (widx >> 5) * 8 + (n >> 3);
        int gw = (widx & 31) * 8 + (n & 7);
        long pbase = (long)b * CHW_ + gh * 256 + gw;
#pragma unroll
        for (int c2 = 0; c2 < 32; ++c2) {
            int col = col0 + wn * 32 + c2;
            outF[pbase + (long)col * HW_] = wbuf[c2 * 33 + lane];
        }
    } else if constexpr (EPI == 3) {
        // lane owns one pixel; 32 contiguous floats per RMW across lanes
        int row = (int)row0 + wm * 32 + lane;
        int b = row >> 16, s = row & (HW_ - 1);
        long pbase = (long)b * CHW_ + s;
#pragma unroll
        for (int c2 = 0; c2 < 32; ++c2) {
            int col = col0 + wn * 32 + c2;
            long addr = pbase + (long)col * HW_;
            outF[addr] += wbuf[c2 * 33 + lane];
        }
    }
}

// ---------------------------------------------------------------------------
// Shifted-window attention: one block = one (window, head); 4 waves, each
// owns a 16-row strip. QK^T (4 WMMA) -> bias+mask+softmax -> PV (4 WMMA).
// ---------------------------------------------------------------------------
__global__ void __launch_bounds__(128)
attn_kernel(const _Float16* __restrict__ qkv, const float* __restrict__ rpb,
            _Float16* __restrict__ attn_out) {
    __shared__ float bias_s[225];
    __shared__ __align__(16) _Float16 probs[64 * 72];

    const int blk = blockIdx.x;                   // 0..12287
    const int bw = blk / 6, head = blk - (blk / 6) * 6;
    const int widx = bw & 1023;
    const int wh = widx >> 5, ww = widx & 31;
    const int tid = threadIdx.x;
    const int wave = tid >> 5, lane = tid & 31;
    const int lh = (lane >> 4) & 1, lm = lane & 15;

    for (int i = tid; i < 225; i += 128) bias_s[i] = rpb[i * 6 + head];
    __syncthreads();

    const _Float16* base = qkv + (long)bw * 64 * 576;
    const int ho = head * 32;

    // S = Q * K^T
    v16h aq = frag_rowA(base + (long)(wave * 16 + lm) * 576 + ho, lh);
    v8f s[4];
#pragma unroll
    for (int t = 0; t < 4; ++t) {
        v8f z = {0.f, 0.f, 0.f, 0.f, 0.f, 0.f, 0.f, 0.f};
        v16h bk = frag_rowB(base + (long)(t * 16 + lm) * 576 + 192 + ho, lh);
        s[t] = wmma_f16(aq, bk, z);
    }

    // scale + rel-pos bias + shift mask + row softmax
    const float scale = 0.17677669529663689f;      // 32^-0.5
#pragma unroll
    for (int rr = 0; rr < 8; ++rr) {
        int i = wave * 16 + rr + lh * 8;
        int yi = i >> 3, xi = i & 7;
        int li = reg8(wh * 8 + yi) * 3 + reg8(ww * 8 + xi);
        float v[4];
        float mx = -3.0e38f;
#pragma unroll
        for (int t = 0; t < 4; ++t) {
            int j = t * 16 + lm;
            int yj = j >> 3, xj = j & 7;
            int lj = reg8(wh * 8 + yj) * 3 + reg8(ww * 8 + xj);
            float bia = bias_s[(yi - yj + 7) * 15 + (xi - xj + 7)];
            v[t] = s[t][rr] * scale + bia + ((li == lj) ? 0.f : -100.f);
            mx = fmaxf(mx, v[t]);
        }
#pragma unroll
        for (int d = 1; d < 16; d <<= 1) mx = fmaxf(mx, __shfl_xor(mx, d, 32));
        float sum = 0.f;
#pragma unroll
        for (int t = 0; t < 4; ++t) { v[t] = __expf(v[t] - mx); sum += v[t]; }
#pragma unroll
        for (int d = 1; d < 16; d <<= 1) sum += __shfl_xor(sum, d, 32);
        float inv = 1.0f / sum;
#pragma unroll
        for (int t = 0; t < 4; ++t)
            probs[i * 72 + t * 16 + lm] = (_Float16)(v[t] * inv);
    }
    __syncthreads();

    // O = P * V
    v8f o0 = {0.f, 0.f, 0.f, 0.f, 0.f, 0.f, 0.f, 0.f};
    v8f o1 = o0;
#pragma unroll
    for (int ks = 0; ks < 2; ++ks) {
        v16h ap = frag_rowA(&probs[(wave * 16 + lm) * 72 + ks * 32], lh);
        v16h bv0 = frag_colB(base + (long)(ks * 32) * 576 + 384 + ho + lm, 576, lh);
        v16h bv1 = frag_colB(base + (long)(ks * 32) * 576 + 384 + ho + 16 + lm, 576, lh);
        o0 = wmma_f16(ap, bv0, o0);
        o1 = wmma_f16(ap, bv1, o1);
    }
#pragma unroll
    for (int rr = 0; rr < 8; ++rr) {
        long row = (long)bw * 64 + wave * 16 + rr + lh * 8;
        attn_out[row * CD_ + ho + lm]      = (_Float16)o0[rr];
        attn_out[row * CD_ + ho + 16 + lm] = (_Float16)o1[rr];
    }
}

// ---------------------------------------------------------------------------
// Global average pool per (b,c)
// ---------------------------------------------------------------------------
__global__ void __launch_bounds__(256)
gap_kernel(const float* __restrict__ y, float* __restrict__ gap) {
    int bc = blockIdx.x;                            // 0..383
    const float* p = y + (long)bc * HW_;
    float sum = 0.f;
    for (int i = threadIdx.x; i < HW_; i += 256) sum += p[i];
    __shared__ float red[256];
    red[threadIdx.x] = sum;
    __syncthreads();
    for (int d = 128; d > 0; d >>= 1) {
        if (threadIdx.x < d) red[threadIdx.x] += red[threadIdx.x + d];
        __syncthreads();
    }
    if (threadIdx.x == 0) gap[bc] = red[0] * (1.f / HW_);
}

// s[b,o] = sum_c gap[b,c]*ca_w[o,c] + ca_b[o]
__global__ void __launch_bounds__(192)
ca_kernel(const float* __restrict__ gap, const float* __restrict__ caw,
          const float* __restrict__ cab, float* __restrict__ scl) {
    int o = threadIdx.x, b = blockIdx.x;
    float s = cab[o];
    for (int c = 0; c < CD_; ++c) s += gap[b * CD_ + c] * caw[o * CD_ + c];
    scl[b * CD_ + o] = s;
}

// out = x + y * s[b,c]
__global__ void __launch_bounds__(256)
residual_kernel(const float* __restrict__ x, const float* __restrict__ y,
                const float* __restrict__ scl, float* __restrict__ out) {
    long i = (long)blockIdx.x * 256 + threadIdx.x;  // < 25,165,824
    int bc = (int)(i >> 16);
    out[i] = x[i] + y[i] * scl[bc];
}

// ---------------------------------------------------------------------------
extern "C" void kernel_launch(void* const* d_in, const int* in_sizes, int n_in,
                              void* d_out, int out_size, void* d_ws, size_t ws_size,
                              hipStream_t stream) {
    (void)in_sizes; (void)n_in; (void)out_size; (void)ws_size;
    const float* x      = (const float*)d_in[0];
    const float* n1w    = (const float*)d_in[1];
    const float* n1b    = (const float*)d_in[2];
    const float* qkv_w  = (const float*)d_in[3];
    const float* qkv_b  = (const float*)d_in[4];
    const float* proj_w = (const float*)d_in[5];
    const float* proj_b = (const float*)d_in[6];
    const float* rpb    = (const float*)d_in[7];
    const float* ca_w   = (const float*)d_in[8];
    const float* ca_b   = (const float*)d_in[9];
    const float* n2w    = (const float*)d_in[10];
    const float* n2b    = (const float*)d_in[11];
    const float* fc1_w  = (const float*)d_in[12];
    const float* fc1_b  = (const float*)d_in[13];
    const float* fc2_w  = (const float*)d_in[14];
    const float* fc2_b  = (const float*)d_in[15];
    float* out = (float*)d_out;

    char* ws = (char*)d_ws;
    _Float16* qkvh = (_Float16*)(ws + OFF_QKV);
    _Float16* regB = (_Float16*)(ws + OFF_B);    // xh -> attn_out -> ln2h
    float*    yat  = (float*)   (ws + OFF_C);    // proj output (BCHW)
    _Float16* hbuf = (_Float16*)(ws + OFF_C);    // gelu(fc1) f16 (reuse)
    float* mu  = (float*)(ws + OFF_MU);
    float* rs  = (float*)(ws + OFF_RS);
    float* gap = (float*)(ws + OFF_GAP);
    float* scl = (float*)(ws + OFF_SCL);
    _Float16* wqh = (_Float16*)(ws + OFF_WQ);
    _Float16* wph = (_Float16*)(ws + OFF_WP);
    _Float16* w1h = (_Float16*)(ws + OFF_W1);
    _Float16* w2h = (_Float16*)(ws + OFF_W2);

    // weights -> f16
    cvt_f16_kernel<<<(110592 + 255) / 256, 256, 0, stream>>>(qkv_w, wqh, 110592);
    cvt_f16_kernel<<<(36864 + 255) / 256, 256, 0, stream>>>(proj_w, wph, 36864);
    cvt_f16_kernel<<<(73728 + 255) / 256, 256, 0, stream>>>(fc1_w, w1h, 73728);
    cvt_f16_kernel<<<(73728 + 255) / 256, 256, 0, stream>>>(fc2_w, w2h, 73728);

    // LN1 + roll + window partition
    ln_stats_kernel<<<512, 256, 0, stream>>>(x, mu, rs);
    ln1_window_kernel<<<98304, 256, 0, stream>>>(x, mu, rs, n1w, n1b, regB);

    // QKV projection
    gemm_kernel<192, 576, 0><<<dim3(1024, 9), 256, 0, stream>>>(
        regB, wqh, qkv_b, nullptr, qkvh);

    // attention (writes attn_out over regB)
    attn_kernel<<<12288, 128, 0, stream>>>(qkvh, rpb, regB);

    // output projection + window_reverse -> y_attn (BCHW f32)
    gemm_kernel<192, 192, 2><<<dim3(1024, 3), 256, 0, stream>>>(
        regB, wph, proj_b, yat, nullptr);

    // channel attention + residual
    gap_kernel<<<384, 256, 0, stream>>>(yat, gap);
    ca_kernel<<<2, 192, 0, stream>>>(gap, ca_w, ca_b, scl);
    residual_kernel<<<98304, 256, 0, stream>>>(x, yat, scl, out);

    // LN2 + MLP (fc1+GELU, fc2 scatter-add into out)
    ln_stats_kernel<<<512, 256, 0, stream>>>(out, mu, rs);
    ln2_apply_kernel<<<98304, 256, 0, stream>>>(out, mu, rs, n2w, n2b, regB);
    gemm_kernel<192, 384, 1><<<dim3(1024, 6), 256, 0, stream>>>(
        regB, w1h, fc1_b, nullptr, hbuf);
    gemm_kernel<384, 192, 3><<<dim3(1024, 3), 256, 0, stream>>>(
        hbuf, w2h, fc2_b, out, nullptr);
}
